// MultiHeadAttention_3856880631829
// MI455X (gfx1250) — matrix operand order repaired and checked by reading, not compiled, not run on hardware
//
#include <hip/hip_runtime.h>
#include <hip/hip_bf16.h>
#include <stdint.h>

// Problem constants (from reference)
static constexpr int S_LEN  = 4096;
static constexpr int BATCH  = 2;
static constexpr int NHEADS = 16;
static constexpr int HDIM   = 64;
static constexpr int EMB    = 1024;
static constexpr int K_IN   = 1024;
static constexpr int N_QKV  = 3072;
static constexpr int WHALF  = 128;

typedef __attribute__((ext_vector_type(16))) __bf16 v16bf;
typedef __attribute__((ext_vector_type(8)))  float  v8f;
typedef __attribute__((ext_vector_type(4)))  unsigned int u32x4;
typedef __attribute__((ext_vector_type(8)))  int i32x8;
typedef __attribute__((ext_vector_type(4)))  int i32x4;

union Frag16 { v16bf v; uint4 q[2]; };

__device__ __forceinline__ unsigned short f2bf_u(float f) {
  union { float f; uint32_t u; } a; a.f = f;
  uint32_t r = a.u + 0x7FFFu + ((a.u >> 16) & 1u);   // round-to-nearest-even
  return (unsigned short)(r >> 16);
}

// Build a 16xbf16 fragment from two contiguous 16-byte chunks (elems 0-7, 8-15)
__device__ __forceinline__ v16bf load_frag(const unsigned short* p0,
                                           const unsigned short* p1) {
  Frag16 f;
  f.q[0] = *reinterpret_cast<const uint4*>(p0);
  f.q[1] = *reinterpret_cast<const uint4*>(p1);
  return f.v;
}

__device__ __forceinline__ v8f wmma_bf16(v16bf a, v16bf b, v8f c) {
  return __builtin_amdgcn_wmma_f32_16x16x32_bf16(false, a, false, b,
                                                 (short)0, c, false, false);
}

__device__ __forceinline__ void wait_ds() {
#if __has_builtin(__builtin_amdgcn_s_wait_dscnt)
  __builtin_amdgcn_s_wait_dscnt(0);
#else
  asm volatile("s_wait_dscnt 0" ::: "memory");
#endif
}

__device__ __forceinline__ void wait_tensor() {
#if __has_builtin(__builtin_amdgcn_s_wait_tensorcnt)
  __builtin_amdgcn_s_wait_tensorcnt(0);
#else
  asm volatile("s_wait_tensorcnt 0" ::: "memory");
#endif
}

// TDM: DMA `n` dwords from global `gp` into LDS offset `lds_off` (1-D tile).
// D# per CDNA5 ISA (group0 + group1; groups 2/3 unused).
__device__ __forceinline__ void tdm_load_dwords_to_lds(const void* gp,
                                                       uint32_t lds_off,
                                                       uint32_t n) {
#if __has_builtin(__builtin_amdgcn_tensor_load_to_lds)
  uint64_t ga = (uint64_t)(uintptr_t)gp;
  u32x4 g0;
  g0[0] = 1u;                                    // count=1, user descriptor
  g0[1] = lds_off;                               // lds_addr
  g0[2] = (uint32_t)ga;                          // global_addr[31:0]
  g0[3] = (uint32_t)((ga >> 32) & 0x01FFFFFFu) | (2u << 30);  // addr hi | type=2
  i32x8 g1;
  g1[0] = (int)(2u << 16);                       // data_size=4B, mask=0
  g1[1] = (int)((n & 0xFFFFu) << 16);            // tensor_dim0[15:0]
  g1[2] = (int)((n >> 16) | (1u << 16));         // dim0 hi | tensor_dim1=1 (lo)
  g1[3] = (int)((n & 0xFFFFu) << 16);            // dim1 hi=0 | tile_dim0=n
  g1[4] = 0;                                     // tile_dim1=0, tile_dim2=0
  g1[5] = (int)n;                                // tensor_dim0_stride[31:0]
  g1[6] = 0;
  g1[7] = 0;
  i32x4 z4 = {0, 0, 0, 0};
#if __clang_major__ >= 23
  i32x8 z8 = {0, 0, 0, 0, 0, 0, 0, 0};
  __builtin_amdgcn_tensor_load_to_lds(g0, g1, z4, z4, z8, 0);
#else
  __builtin_amdgcn_tensor_load_to_lds(g0, g1, z4, z4, 0);
#endif
#else
  (void)gp; (void)lds_off; (void)n;
#endif
}

// ---------------------------------------------------------------------------
// f32 -> bf16 conversion (n multiple of 1024; each thread does 4)
// ---------------------------------------------------------------------------
__global__ void cvt_f32_to_bf16(const float* __restrict__ src,
                                unsigned short* __restrict__ dst, int n) {
  int i = (blockIdx.x * blockDim.x + threadIdx.x) * 4;
  if (i + 3 < n) {
    float4 v = *reinterpret_cast<const float4*>(src + i);
    ushort4 o;
    o.x = f2bf_u(v.x); o.y = f2bf_u(v.y); o.z = f2bf_u(v.z); o.w = f2bf_u(v.w);
    *reinterpret_cast<ushort4*>(dst + i) = o;
  }
}

// ---------------------------------------------------------------------------
// Transpose f32 [R x C] -> bf16 [C x R] (32x32 LDS tiles; R,C multiples of 32)
// ---------------------------------------------------------------------------
__global__ __launch_bounds__(256)
void transpose_f32_to_bf16(const float* __restrict__ src,
                           unsigned short* __restrict__ dst, int R, int C) {
  __shared__ unsigned short tile[32][33];
  const int c0 = blockIdx.x * 32, r0 = blockIdx.y * 32;
  const int tx = threadIdx.x, ty = threadIdx.y;   // (32, 8)
  #pragma unroll
  for (int i = 0; i < 4; i++) {
    int r = r0 + ty + i * 8;
    tile[ty + i * 8][tx] = f2bf_u(src[(size_t)r * C + c0 + tx]);
  }
  __syncthreads();
  #pragma unroll
  for (int i = 0; i < 4; i++) {
    int c = c0 + ty + i * 8;
    dst[(size_t)c * R + r0 + tx] = tile[tx][ty + i * 8];
  }
}

// ---------------------------------------------------------------------------
// LDS-free bf16 WMMA GEMM (weights pre-transposed, L2-resident):
//   C[M,N] = A[M,K] * BT[N,K]^T + bias
// Block 256 threads = 8 waves (4x2), block tile 256x128, wave tile 64x64,
// 16 wmma per K-step of 32. Bias tile DMA'd to LDS by the Tensor Data Mover,
// overlapped with the K-loop.
//   MODE 0: f32 row-major store
//   MODE 1: QKV epilogue -> Qb (scaled 1/8), Kb, V-transposed (all bf16)
// ---------------------------------------------------------------------------
template<int MODE>
__global__ __launch_bounds__(256)
void gemm_bf16_wmma(const unsigned short* __restrict__ A,
                    const unsigned short* __restrict__ BT,
                    const float* __restrict__ bias,
                    float* __restrict__ Cout,
                    unsigned short* __restrict__ Qb,
                    unsigned short* __restrict__ Kb,
                    unsigned short* __restrict__ VT,
                    int M, int N, int K) {
  __shared__ __align__(16) float biasLds[128];

  const int t   = threadIdx.x;
  const int wid = t >> 5, lane = t & 31;
  const int hf  = lane >> 4, l15 = lane & 15;
  const int wm  = wid >> 1, wn = wid & 1;          // 4x2 wave grid
  const int m0  = blockIdx.y * 256, n0 = blockIdx.x * 128;

#if __has_builtin(__builtin_amdgcn_tensor_load_to_lds)
  if (wid == 0) {
    uint32_t lds_off = (uint32_t)(uintptr_t)(void*)biasLds;
    tdm_load_dwords_to_lds(bias + n0, lds_off, 128u);
  }
#else
  if (t < 128) biasLds[t] = bias[n0 + t];
#endif

  v8f acc[4][4];
  #pragma unroll
  for (int mm = 0; mm < 4; mm++)
    #pragma unroll
    for (int nn = 0; nn < 4; nn++) {
      v8f z = {0.f,0.f,0.f,0.f,0.f,0.f,0.f,0.f};
      acc[mm][nn] = z;
    }

  const unsigned short* aBase = A + (size_t)(m0 + wm * 64 + l15) * K;
  const unsigned short* bBase = BT + (size_t)(n0 + wn * 64 + l15) * K;

  for (int kt = 0; kt < K; kt += 32) {
    v16bf af[4], bfv[4];
    #pragma unroll
    for (int mm = 0; mm < 4; mm++) {
      const unsigned short* p = aBase + (size_t)(mm * 16) * K + kt;
      af[mm] = load_frag(p + 16 * hf, p + 16 * hf + 8);      // A 16x32 frag
    }
    #pragma unroll
    for (int nn = 0; nn < 4; nn++) {
      const unsigned short* p = bBase + (size_t)(nn * 16) * K + kt;
      bfv[nn] = load_frag(p + 16 * hf, p + 16 * hf + 8);     // B 32x16 frag
    }
    #pragma unroll
    for (int mm = 0; mm < 4; mm++)
      #pragma unroll
      for (int nn = 0; nn < 4; nn++)
        acc[mm][nn] = wmma_bf16(af[mm], bfv[nn], acc[mm][nn]);
  }

#if __has_builtin(__builtin_amdgcn_tensor_load_to_lds)
  if (wid == 0) wait_tensor();
#endif
  __syncthreads();

  // Epilogue. C layout: row = r + 8*hf, col = lane&15 within each 16x16 tile.
  #pragma unroll
  for (int mm = 0; mm < 4; mm++) {
    #pragma unroll
    for (int nn = 0; nn < 4; nn++) {
      #pragma unroll
      for (int r = 0; r < 8; r++) {
        int grow = m0 + wm * 64 + mm * 16 + r + 8 * hf;
        int gcol = n0 + wn * 64 + nn * 16 + l15;
        float val = acc[mm][nn][r] + biasLds[gcol - n0];
        if (MODE == 0) {
          Cout[(size_t)grow * N + gcol] = val;
        } else {
          int b = grow >> 12, s = grow & (S_LEN - 1);
          int h = gcol / 192, rem = gcol - h * 192;
          int d = rem & 63;
          size_t hb = (size_t)(b * NHEADS + h);
          if (rem < 64)
            Qb[(hb * S_LEN + s) * HDIM + d] = f2bf_u(val * 0.125f);  // 1/sqrt(64)
          else if (rem < 128)
            Kb[(hb * S_LEN + s) * HDIM + d] = f2bf_u(val);
          else
            VT[(hb * HDIM + d) * S_LEN + s] = f2bf_u(val);           // transposed
        }
      }
    }
  }
}

// ---------------------------------------------------------------------------
// Banded flash attention. One wave per 16-query tile; 9 chunks of 32 keys
// cover the [q0-128, q0+143] union window.
// ---------------------------------------------------------------------------
__global__ __launch_bounds__(128)
void band_attn(const unsigned short* __restrict__ Qb,
               const unsigned short* __restrict__ Kb,
               const unsigned short* __restrict__ VT,
               const int* __restrict__ pm,
               unsigned short* __restrict__ attn) {
  __shared__ __align__(16) unsigned short Ps[4][16 * 40];  // per-wave P staging

  const int t   = threadIdx.x;
  const int wid = t >> 5, lane = t & 31;
  const int hf  = lane >> 4, l15 = lane & 15;
  const int h   = blockIdx.y, b = blockIdx.z;
  const int q0  = (blockIdx.x * 4 + wid) * 16;
  const size_t bh = (size_t)(b * NHEADS + h);

  // Q fragments (A-layout): row = l15, two d-chunks of 32
  v16bf qf[2];
  {
    const unsigned short* qr = Qb + (bh * S_LEN + (q0 + l15)) * HDIM;
    #pragma unroll
    for (int dc = 0; dc < 2; dc++)
      qf[dc] = load_frag(qr + dc * 32 + 16 * hf, qr + dc * 32 + 16 * hf + 8);
  }

  v8f o[4];
  #pragma unroll
  for (int nn = 0; nn < 4; nn++) {
    v8f z = {0.f,0.f,0.f,0.f,0.f,0.f,0.f,0.f};
    o[nn] = z;
  }
  float mrow[8], lrow[8];
  #pragma unroll
  for (int r = 0; r < 8; r++) { mrow[r] = -3.0e38f; lrow[r] = 0.f; }

  for (int cc = 0; cc < 9; cc++) {
    const int jc = q0 - WHALF + cc * 32;

    // --- scores: two 16-key subtiles, each = 2 chained wmma over d ---
    v8f st[2];
    #pragma unroll
    for (int s2 = 0; s2 < 2; s2++) {
      int key = jc + s2 * 16 + l15;
      int kcl = key < 0 ? 0 : (key > S_LEN - 1 ? S_LEN - 1 : key);
      const unsigned short* kr = Kb + (bh * S_LEN + kcl) * HDIM;
      v16bf kf0 = load_frag(kr + 16 * hf, kr + 16 * hf + 8);           // d 0..31
      v16bf kf1 = load_frag(kr + 32 + 16 * hf, kr + 32 + 16 * hf + 8); // d 32..63
      v8f z = {0.f,0.f,0.f,0.f,0.f,0.f,0.f,0.f};
      z = wmma_bf16(qf[0], kf0, z);
      z = wmma_bf16(qf[1], kf1, z);
      st[s2] = z;
    }

    // --- band / bounds / padding mask (Q pre-scaled, so scores are scaled) ---
    int j0 = jc + l15, j1 = jc + 16 + l15;
    int j0c = j0 < 0 ? 0 : (j0 > S_LEN - 1 ? S_LEN - 1 : j0);
    int j1c = j1 < 0 ? 0 : (j1 > S_LEN - 1 ? S_LEN - 1 : j1);
    bool p0ok = (j0 >= 0) && (j0 < S_LEN) && (pm[b * S_LEN + j0c] != 0);
    bool p1ok = (j1 >= 0) && (j1 < S_LEN) && (pm[b * S_LEN + j1c] != 0);
    #pragma unroll
    for (int r = 0; r < 8; r++) {
      int i = q0 + r + 8 * hf;
      bool v0 = p0ok && (j0 >= i - WHALF) && (j0 <= i + WHALF);
      bool v1 = p1ok && (j1 >= i - WHALF) && (j1 <= i + WHALF);
      st[0][r] = v0 ? st[0][r] : -1.0e15f;
      st[1][r] = v1 ? st[1][r] : -1.0e15f;
    }

    // --- online softmax update; stage P (bf16) into LDS in A-layout order ---
    #pragma unroll
    for (int r = 0; r < 8; r++) {
      float mx = fmaxf(st[0][r], st[1][r]);
      #pragma unroll
      for (int off = 1; off < 16; off <<= 1)
        mx = fmaxf(mx, __shfl_xor(mx, off, 32));
      float mnew  = fmaxf(mrow[r], mx);
      float alpha = __expf(mrow[r] - mnew);
      float p0 = __expf(st[0][r] - mnew);
      float p1 = __expf(st[1][r] - mnew);
      float rs = p0 + p1;
      #pragma unroll
      for (int off = 1; off < 16; off <<= 1)
        rs += __shfl_xor(rs, off, 32);
      lrow[r] = lrow[r] * alpha + rs;
      mrow[r] = mnew;
      #pragma unroll
      for (int nn = 0; nn < 4; nn++) o[nn][r] *= alpha;
      int row = r + 8 * hf;
      Ps[wid][row * 40 + l15]      = f2bf_u(p0);
      Ps[wid][row * 40 + 16 + l15] = f2bf_u(p1);
    }
    wait_ds();  // LDS store -> load fence (same wave, split DS counter)

    // P as A-fragment (16x32)
    v16bf pf = load_frag(&Ps[wid][l15 * 40 + 16 * hf],
                         &Ps[wid][l15 * 40 + 16 * hf + 8]);

    // --- V fragments from V-transposed; clamp fully-OOB subtiles (P cols=0) ---
    int base0 = (jc >= 0 && jc + 16 <= S_LEN) ? jc : 0;
    int base1 = (jc + 16 >= 0 && jc + 32 <= S_LEN) ? jc + 16 : 0;
    int basesel = hf ? base1 : base0;
    #pragma unroll
    for (int nn = 0; nn < 4; nn++) {
      const unsigned short* vr =
          VT + (bh * HDIM + nn * 16 + l15) * S_LEN + basesel;
      v16bf vf = load_frag(vr, vr + 8);
      o[nn] = wmma_bf16(pf, vf, o[nn]);
    }
  }

  // --- normalize and store (bf16 activations for the output GEMM) ---
  #pragma unroll
  for (int r = 0; r < 8; r++) {
    float inv = 1.0f / lrow[r];
    int row = q0 + r + 8 * hf;
    #pragma unroll
    for (int nn = 0; nn < 4; nn++) {
      int col = h * HDIM + nn * 16 + l15;
      attn[((size_t)b * S_LEN + row) * EMB + col] = f2bf_u(o[nn][r] * inv);
    }
  }
}

// ---------------------------------------------------------------------------
extern "C" void kernel_launch(void* const* d_in, const int* in_sizes, int n_in,
                              void* d_out, int out_size, void* d_ws, size_t ws_size,
                              hipStream_t stream) {
  const float* x     = (const float*)d_in[0];
  const int*   pmask = (const int*)d_in[1];
  const float* Wqkv  = (const float*)d_in[2];
  const float* bqkv  = (const float*)d_in[3];
  const float* Wo    = (const float*)d_in[4];
  const float* bo    = (const float*)d_in[5];
  float* out = (float*)d_out;

  char* ws = (char*)d_ws;
  size_t off = 0;
  auto alloc = [&](size_t bytes) -> char* {
    char* p = ws + off;
    off += (bytes + 255) & ~(size_t)255;
    return p;
  };
  const int M = BATCH * S_LEN;  // 8192
  unsigned short* x_bf    = (unsigned short*)alloc((size_t)M * K_IN * 2);
  unsigned short* wqkvT   = (unsigned short*)alloc((size_t)K_IN * N_QKV * 2);
  unsigned short* woT     = (unsigned short*)alloc((size_t)EMB * EMB * 2);
  unsigned short* Qb      = (unsigned short*)alloc((size_t)M * EMB * 2);
  unsigned short* Kb      = (unsigned short*)alloc((size_t)M * EMB * 2);
  unsigned short* VT      = (unsigned short*)alloc((size_t)M * EMB * 2);
  unsigned short* attnb   = (unsigned short*)alloc((size_t)M * EMB * 2);
  (void)ws_size; (void)in_sizes; (void)n_in; (void)out_size;

  int n1 = M * K_IN;
  cvt_f32_to_bf16<<<n1 / 1024, 256, 0, stream>>>(x, x_bf, n1);

  // Weights -> bf16, transposed once (L2-resident thereafter)
  transpose_f32_to_bf16<<<dim3(N_QKV / 32, K_IN / 32), dim3(32, 8), 0, stream>>>(
      Wqkv, wqkvT, K_IN, N_QKV);
  transpose_f32_to_bf16<<<dim3(EMB / 32, EMB / 32), dim3(32, 8), 0, stream>>>(
      Wo, woT, EMB, EMB);

  dim3 g1(N_QKV / 128, M / 256);
  gemm_bf16_wmma<1><<<g1, 256, 0, stream>>>(x_bf, wqkvT, bqkv, nullptr,
                                            Qb, Kb, VT, M, N_QKV, K_IN);

  dim3 g2(S_LEN / 64, NHEADS, BATCH);
  band_attn<<<g2, 128, 0, stream>>>(Qb, Kb, VT, pmask, attnb);

  dim3 g3(EMB / 128, M / 256);
  gemm_bf16_wmma<0><<<g3, 256, 0, stream>>>(attnb, woT, bo, out,
                                            nullptr, nullptr, nullptr,
                                            M, EMB, EMB);
}
